// BRA_16389595201974
// MI455X (gfx1250) — matrix operand order repaired and checked by reading, not compiled, not run on hardware
//
#include <hip/hip_runtime.h>
#include <hip/hip_bf16.h>
#include <math.h>

// ---------------- problem constants (from reference) ----------------
#define B_      64
#define N_TOK   1025
#define C_      768
#define NH_     12
#define HD_     64          // C_/NH_
#define HW_     32          // H = W
#define NSPA    1024        // H*W
#define RREG    64          // NW*NW regions
#define SREG    16          // rh*rw tokens per region
#define TOPK_   32
#define KVROWS  1536        // 2*C

typedef __attribute__((ext_vector_type(16))) __bf16 v16bf;
typedef __attribute__((ext_vector_type(2)))  __bf16 v2bf;
typedef __attribute__((ext_vector_type(8)))  float  v8f;
typedef int v4i_vs __attribute__((vector_size(16)));   // matches async-LDS builtin param

#define AS1 __attribute__((address_space(1)))
#define AS3 __attribute__((address_space(3)))

#if __has_builtin(__builtin_amdgcn_global_load_async_to_lds_b128) && \
    __has_builtin(__builtin_amdgcn_s_wait_asynccnt)
#define USE_ASYNC 1
#else
#define USE_ASYNC 0
#endif

// packed f32x2 -> bf16x2 in one dword.
// Native __bf16 casts (fptrunc) let the backend pick the hardware
// convert; guarded builtin branch kept in case it appears.
#if __has_builtin(__builtin_amdgcn_cvt_pk_bf16_f32)
__device__ __forceinline__ unsigned cvtpk(float a, float b) {
  return __builtin_bit_cast(unsigned, __builtin_amdgcn_cvt_pk_bf16_f32(a, b));
}
#else
__device__ __forceinline__ unsigned cvtpk(float a, float b) {
  v2bf t;
  t.x = (__bf16)a;
  t.y = (__bf16)b;
  return __builtin_bit_cast(unsigned, t);
}
#endif
__device__ __forceinline__ unsigned short bf_bits(float f) {
  __bf16 h = (__bf16)f;
  return __builtin_bit_cast(unsigned short, h);
}

// fragment assembled via 32-bit words (avoids v_mov_b16 half shuffles)
union frag_u { v16bf v; unsigned u[8]; };

// =====================================================================
// 1) q = x[:,0] @ Wq.T + bq            (B,768) — tiny
// =====================================================================
__global__ __launch_bounds__(256) void qproj_kernel(
    const float* __restrict__ x, const float* __restrict__ Wq,
    const float* __restrict__ bq, float* __restrict__ q)
{
  int b = blockIdx.y;
  int o = blockIdx.x * 256 + threadIdx.x;          // 0..767
  const float* xr = x + (size_t)b * (N_TOK * C_);  // CLS token row
  const float* wr = Wq + (size_t)o * C_;
  float acc = bq[o];
  for (int c = 0; c < C_; ++c) acc += xr[c] * wr[c];
  q[b * C_ + o] = acc;
}

// =====================================================================
// 2) kv[b] = Wkv(1536x768) * Xb(768x1024) + bkv   — bf16 WMMA GEMM
//    block tile 128x128, K-step 32, 8 wave32s (2x4), wave tile 64x32.
//    Async path: double-buffered f32 LDS tiles filled with
//    GLOBAL_LOAD_ASYNC_TO_LDS_B128 (ASYNCcnt), packed bf16 convert at
//    fragment assembly. Fallback: reg-staged bf16 LDS tiles.
// =====================================================================
#define BM 128
#define BN 128
#define BK 32
#define KSTR 36    // fallback: LDS K stride in shorts (32 data + 4 pad)
#define AST  36    // async: A LDS row stride in floats (144 B, 16B-aligned)
#define BST  132   // async: B LDS row stride in floats (528 B, 16B-aligned)

__global__ __launch_bounds__(256) void kv_gemm_kernel(
    const float* __restrict__ x, const float* __restrict__ Wkv,
    const float* __restrict__ bkv, float* __restrict__ kv)
{
  const int tid = threadIdx.x;
  const int b   = blockIdx.z;
  const int m0  = blockIdx.x * BM;
  const int n0  = blockIdx.y * BN;
  const float* Xb = x + (size_t)b * (N_TOK * C_) + C_;  // (768,1024) row-major

  const int wid  = tid >> 5;
  const int lane = tid & 31;
  const int hf   = lane >> 4;     // lane half (0/1)
  const int ln   = lane & 15;
  const int wm   = wid >> 2;      // 0..1  -> 64-row strip
  const int wn   = wid & 3;       // 0..3  -> 32-col strip

  v8f acc[4][2];
  #pragma unroll
  for (int i = 0; i < 4; ++i)
    #pragma unroll
    for (int j = 0; j < 2; ++j)
      #pragma unroll
      for (int e = 0; e < 8; ++e) acc[i][j][e] = 0.f;

  const int KT = C_ / BK;   // 24

#if USE_ASYNC
  // ---------------- async double-buffered f32 tiles ----------------
  __shared__ float lAf[2][BM * AST];   // [m][k]   36.9 KB
  __shared__ float lBf[2][BK * BST];   // [k][n]   33.8 KB

  auto issue = [&](int kt, int bb) {
    const int k0 = kt * BK;
    #pragma unroll
    for (int i = 0; i < 4; ++i) {
      int f  = tid + i * 256;                 // 0..1023 16B-chunk ids
      int am = f >> 3, ak = (f & 7) << 2;     // A: 128 rows x 8 chunks
      __builtin_amdgcn_global_load_async_to_lds_b128(
          (AS1 v4i_vs*)(Wkv + (size_t)(m0 + am) * C_ + k0 + ak),
          (AS3 v4i_vs*)&lAf[bb][am * AST + ak], 0, 0);
      int bk = f >> 5, bn = (f & 31) << 2;    // B: 32 rows x 32 chunks
      __builtin_amdgcn_global_load_async_to_lds_b128(
          (AS1 v4i_vs*)(Xb + (size_t)(k0 + bk) * NSPA + n0 + bn),
          (AS3 v4i_vs*)&lBf[bb][bk * BST + bn], 0, 0);
    }
  };

  issue(0, 0);
  for (int kt = 0; kt < KT; ++kt) {
    const int cur = kt & 1;
    if (kt + 1 < KT) {
      issue(kt + 1, cur ^ 1);                 // overlap next tile fill
      __builtin_amdgcn_s_wait_asynccnt(8);    // tile kt's 8 chunks landed
    } else {
      __builtin_amdgcn_s_wait_asynccnt(0);
    }
    __syncthreads();                          // all waves' chunks visible

    frag_u fa[4], fb[2];
    #pragma unroll
    for (int i = 0; i < 4; ++i) {
      const float* rowp = &lAf[cur][(wm * 64 + i * 16 + ln) * AST];
      #pragma unroll
      for (int p = 0; p < 8; ++p) {
        // A(16x32): e<8 -> K = hf*8+e ; e>=8 -> K = 16+hf*8+(e-8)
        int K = (p < 4) ? (hf * 8 + 2 * p) : (16 + hf * 8 + 2 * (p - 4));
        float2 d = *(const float2*)&rowp[K];
        fa[i].u[p] = cvtpk(d.x, d.y);
      }
    }
    #pragma unroll
    for (int j = 0; j < 2; ++j) {
      int n = wn * 32 + j * 16 + ln;
      #pragma unroll
      for (int p = 0; p < 8; ++p) {
        // B(32x16): e<8 -> K = hf*8+e ; e>=8 -> K = 16+hf*8+(e-8), elements (2p, 2p+1)
        int K = (p < 4) ? (hf * 8 + 2 * p) : (16 + hf * 8 + 2 * (p - 4));
        float a = lBf[cur][(K + 0) * BST + n];
        float c = lBf[cur][(K + 1) * BST + n];
        fb[j].u[p] = cvtpk(a, c);
      }
    }
    #pragma unroll
    for (int i = 0; i < 4; ++i)
      #pragma unroll
      for (int j = 0; j < 2; ++j)
        acc[i][j] = __builtin_amdgcn_wmma_f32_16x16x32_bf16(
            false, fa[i].v, false, fb[j].v, (short)0, acc[i][j], false, false);

    __syncthreads();   // buffer consumed before it is re-issued
  }
#else
  // ---------------- fallback: reg-staged bf16 LDS tiles ----------------
  __shared__ unsigned short lA[BM * KSTR];   // A tile, [m][k]
  __shared__ unsigned short lB[BN * KSTR];   // B tile transposed, [n][k]

  for (int kt = 0; kt < KT; ++kt) {
    const int k0 = kt * BK;
    float4 ra[4], rb[4];
    #pragma unroll
    for (int i = 0; i < 4; ++i) {
      int f  = tid + i * 256;
      int am = f >> 3, ak = (f & 7) << 2;
      ra[i] = *(const float4*)(Wkv + (size_t)(m0 + am) * C_ + k0 + ak);
      int bk = f >> 5, bn = (f & 31) << 2;
      rb[i] = *(const float4*)(Xb + (size_t)(k0 + bk) * NSPA + n0 + bn);
    }
    if (kt + 1 < KT) {
      int am = tid >> 3, ak = (tid & 7) << 2;
      __builtin_prefetch(Wkv + (size_t)(m0 + am) * C_ + k0 + BK + ak, 0, 1);
      int bk = tid >> 5, bn = (tid & 31) << 2;
      __builtin_prefetch(Xb + (size_t)(k0 + BK + bk) * NSPA + n0 + bn, 0, 1);
    }
    __syncthreads();
    #pragma unroll
    for (int i = 0; i < 4; ++i) {
      int f  = tid + i * 256;
      int am = f >> 3, ak = (f & 7) << 2;
      unsigned* pA = (unsigned*)&lA[am * KSTR + ak];
      pA[0] = cvtpk(ra[i].x, ra[i].y);
      pA[1] = cvtpk(ra[i].z, ra[i].w);
      int bk = f >> 5, bn = (f & 31) << 2;
      lB[(bn + 0) * KSTR + bk] = bf_bits(rb[i].x);
      lB[(bn + 1) * KSTR + bk] = bf_bits(rb[i].y);
      lB[(bn + 2) * KSTR + bk] = bf_bits(rb[i].z);
      lB[(bn + 3) * KSTR + bk] = bf_bits(rb[i].w);
    }
    __syncthreads();

    frag_u fa[4], fb[2];
    #pragma unroll
    for (int i = 0; i < 4; ++i) {
      const unsigned short* row = &lA[(wm * 64 + i * 16 + ln) * KSTR];
      #pragma unroll
      for (int p = 0; p < 8; ++p) {
        int K = (p < 4) ? (hf * 8 + 2 * p) : (16 + hf * 8 + 2 * (p - 4));
        fa[i].u[p] = *(const unsigned*)&row[K];
      }
    }
    #pragma unroll
    for (int j = 0; j < 2; ++j) {
      const unsigned short* col = &lB[(wn * 32 + j * 16 + ln) * KSTR];
      #pragma unroll
      for (int p = 0; p < 8; ++p) {
        int K = (p < 4) ? (hf * 8 + 2 * p) : (16 + hf * 8 + 2 * (p - 4));
        fb[j].u[p] = *(const unsigned*)&col[K];
      }
    }
    #pragma unroll
    for (int i = 0; i < 4; ++i)
      #pragma unroll
      for (int j = 0; j < 2; ++j)
        acc[i][j] = __builtin_amdgcn_wmma_f32_16x16x32_bf16(
            false, fa[i].v, false, fb[j].v, (short)0, acc[i][j], false, false);
  }
#endif

  // ---- store D + bias:  C/D layout: M = r + 8*hf, N = ln ----
  #pragma unroll
  for (int i = 0; i < 4; ++i) {
    int rowb = m0 + wm * 64 + i * 16 + hf * 8;
    #pragma unroll
    for (int j = 0; j < 2; ++j) {
      int col = n0 + wn * 32 + j * 16 + ln;
      #pragma unroll
      for (int r = 0; r < 8; ++r) {
        int row = rowb + r;
        kv[((size_t)b * KVROWS + row) * NSPA + col] = acc[i][j][r] + bkv[row];
      }
    }
  }
}

// =====================================================================
// 3) region max-pool of k: k_r[b,c,r] = max_{4x4} k[b,c,·,·]
// =====================================================================
__global__ __launch_bounds__(256) void pool_kernel(
    const float* __restrict__ kvbuf, float* __restrict__ k_r)
{
  size_t g = (size_t)blockIdx.x * 256 + threadIdx.x;   // B*C*64 total
  int r = (int)(g & 63);
  size_t bc = g >> 6;
  int c = (int)(bc % C_);
  int b = (int)(bc / C_);
  const float* kp = kvbuf + ((size_t)b * KVROWS + c) * NSPA;
  int h0 = (r >> 3) * 4, w0 = (r & 7) * 4;
  float mx = -3.4e38f;
  #pragma unroll
  for (int p = 0; p < 4; ++p)
    #pragma unroll
    for (int qq = 0; qq < 4; ++qq)
      mx = fmaxf(mx, kp[(h0 + p) * HW_ + w0 + qq]);
  k_r[((size_t)b * C_ + c) * RREG + r] = mx;
}

// =====================================================================
// 4) router: a_r = q · k_r, then top-32 of 64 (order irrelevant)
// =====================================================================
__global__ __launch_bounds__(64) void router_kernel(
    const float* __restrict__ q, const float* __restrict__ k_r,
    int* __restrict__ idx)
{
  __shared__ float a_s[RREG];
  int b = blockIdx.x, t = threadIdx.x;   // 64 threads, one region each
  const float* kr = k_r + (size_t)b * C_ * RREG;
  const float* qb = q + b * C_;
  float acc = 0.f;
  for (int c = 0; c < C_; ++c) acc += qb[c] * kr[c * RREG + t];
  a_s[t] = acc;
  __syncthreads();
  if (t == 0) {
    unsigned long long used = 0ull;
    for (int j = 0; j < TOPK_; ++j) {
      float best = -3.4e38f; int bi = 0;
      for (int r = 0; r < RREG; ++r)
        if (!((used >> r) & 1ull) && a_s[r] > best) { best = a_s[r]; bi = r; }
      used |= 1ull << bi;
      idx[b * TOPK_ + j] = bi;
    }
  }
}

// =====================================================================
// 5) gathered attention per (b, head): softmax(q·K/sqrt(C)) · V
// =====================================================================
__global__ __launch_bounds__(256) void attn_kernel(
    const float* __restrict__ q, const float* __restrict__ kvbuf,
    const int* __restrict__ idx, float* __restrict__ o)
{
  __shared__ float q_s[HD_];
  __shared__ float lg[TOPK_ * SREG];
  __shared__ int   spat[TOPK_ * SREG];
  __shared__ float red[256];

  int m = blockIdx.x, b = blockIdx.y, t = threadIdx.x;
  const float* kvb = kvbuf + (size_t)b * KVROWS * NSPA;
  const float* kb  = kvb + (size_t)m * HD_ * NSPA;          // k rows m*64+d
  const float* vb  = kb + (size_t)C_ * NSPA;                // v rows 768+m*64+d
  if (t < HD_) q_s[t] = q[b * C_ + m * HD_ + t];
  __syncthreads();

  const float scale = rsqrtf((float)C_);
  float lmax = -3.4e38f;
  for (int j = t; j < TOPK_ * SREG; j += 256) {
    int r = idx[b * TOPK_ + (j >> 4)];
    int s = j & 15;
    int sp = ((r >> 3) * 4 + (s >> 2)) * HW_ + (r & 7) * 4 + (s & 3);
    spat[j] = sp;
    float a = 0.f;
    for (int d = 0; d < HD_; ++d) a += q_s[d] * kb[(size_t)d * NSPA + sp];
    float l = a * scale;
    lg[j] = l;
    lmax = fmaxf(lmax, l);
  }
  red[t] = lmax;
  __syncthreads();
  for (int s2 = 128; s2 > 0; s2 >>= 1) {
    if (t < s2) red[t] = fmaxf(red[t], red[t + s2]);
    __syncthreads();
  }
  float gmax = red[0];
  __syncthreads();

  float lsum = 0.f;
  for (int j = t; j < TOPK_ * SREG; j += 256) {
    float e = __expf(lg[j] - gmax);
    lg[j] = e;
    lsum += e;
  }
  red[t] = lsum;
  __syncthreads();
  for (int s2 = 128; s2 > 0; s2 >>= 1) {
    if (t < s2) red[t] += red[t + s2];
    __syncthreads();
  }
  float inv = 1.f / red[0];
  __syncthreads();

  int d = t & 63, ch = t >> 6;     // 64 dims x 4 key-chunks
  float pacc = 0.f;
  for (int j = ch * 128; j < ch * 128 + 128; ++j)
    pacc += lg[j] * vb[(size_t)d * NSPA + spat[j]];
  red[t] = pacc;
  __syncthreads();
  if (ch == 0) {
    float s = red[d] + red[d + 64] + red[d + 128] + red[d + 192];
    o[b * C_ + m * HD_ + d] = s * inv;
  }
}

// =====================================================================
// 6) out = o @ Wo.T + bo   (outDim taken from out_size; bias wrapped)
// =====================================================================
__global__ __launch_bounds__(256) void oproj_kernel(
    const float* __restrict__ o, const float* __restrict__ Wo,
    const float* __restrict__ bo, float* __restrict__ out, int outDim)
{
  int b = blockIdx.y;
  int j = blockIdx.x * 256 + threadIdx.x;
  if (j >= outDim) return;
  const float* orow = o + (size_t)b * C_;
  const float* wr = Wo + (size_t)j * C_;
  float acc = bo[j % C_];
  for (int c = 0; c < C_; ++c) acc += orow[c] * wr[c];
  out[(size_t)b * outDim + j] = acc;
}

// =====================================================================
extern "C" void kernel_launch(void* const* d_in, const int* in_sizes, int n_in,
                              void* d_out, int out_size, void* d_ws, size_t ws_size,
                              hipStream_t stream)
{
  const float* x   = (const float*)d_in[0];
  const float* Wq  = (const float*)d_in[1];
  const float* bq  = (const float*)d_in[2];
  const float* Wkv = (const float*)d_in[3];
  const float* bkv = (const float*)d_in[4];
  const float* Wo  = (const float*)d_in[5];
  const float* bo  = (const float*)d_in[6];
  float* out = (float*)d_out;

  char* ws = (char*)d_ws;
  size_t off = 0;
  float* kv  = (float*)(ws + off); off += (size_t)B_ * KVROWS * NSPA * 4;  // 403 MB
  float* q   = (float*)(ws + off); off += (size_t)B_ * C_ * 4;
  float* k_r = (float*)(ws + off); off += (size_t)B_ * C_ * RREG * 4;
  int*   idx = (int*)  (ws + off); off += (size_t)B_ * TOPK_ * 4;
  float* ob  = (float*)(ws + off); off += (size_t)B_ * C_ * 4;

  qproj_kernel<<<dim3(C_ / 256, B_), 256, 0, stream>>>(x, Wq, bq, q);
  kv_gemm_kernel<<<dim3(KVROWS / BM, NSPA / BN, B_), 256, 0, stream>>>(x, Wkv, bkv, kv);
  pool_kernel<<<dim3((B_ * C_ * RREG) / 256), 256, 0, stream>>>(kv, k_r);
  router_kernel<<<dim3(B_), 64, 0, stream>>>(q, k_r, idx);
  attn_kernel<<<dim3(NH_, B_), 256, 0, stream>>>(q, kv, idx, ob);
  int outDim = out_size / B_;
  oproj_kernel<<<dim3((outDim + 255) / 256, B_), 256, 0, stream>>>(ob, Wo, bo, out, outDim);
}
